// TemplatePointwiseAttention_60773787238711
// MI455X (gfx1250) — compile-verified, matrix operand order checked by reading, not run on hardware
//
#include <hip/hip_runtime.h>

#define CT 64
#define CZ 128
#define CHID 16
#define NHEADS 4
#define NTEMPL 4
#define NRES 512
#define PIXELS (NRES*NRES)
#define NTILES (PIXELS/16)
#define INF_F 1e5f
#define LOG2E 1.44269504088896f

typedef __bf16 v16bf __attribute__((ext_vector_type(16)));
typedef float  v8f   __attribute__((ext_vector_type(8)));

union BFU { __bf16 h[16]; unsigned int d[8]; v16bf v; };

__device__ __forceinline__ unsigned int pack2bf(float lo, float hi) {
  union { __bf16 h[2]; unsigned int d; } u;
  u.h[0] = (__bf16)lo;   // native v_cvt_pk_bf16_f32
  u.h[1] = (__bf16)hi;
  return u.d;
}

// Weight B-fragment ids in LDS (each fragment = 32 lanes * 8 dwords = 1KB)
#define FRAG_WQ(c,n)  ((c)*4+(n))        // c in 0..3 (K chunks of 32, K=128), n in 0..3 (N=64)
#define FRAG_WK(c,n)  (16+(c)*4+(n))     // c in 0..1 (K=64), n in 0..3
#define FRAG_WV(c,n)  (24+(c)*4+(n))
#define FRAG_WO(c,n)  (32+(c)*8+(n))     // c in 0..1 (K=64), n in 0..7 (N=128)
#define NFRAGS 48

__device__ __forceinline__ v8f wmma_bf16(v16bf a, v16bf b, v8f c) {
  return __builtin_amdgcn_wmma_f32_16x16x32_bf16(false, a, false, b, (short)0, c,
                                                 false, false);
}

// Load a pre-swizzled B fragment from LDS: lane reads 8 consecutive dwords.
__device__ __forceinline__ v16bf load_bfrag(const unsigned int* fragLDS, int f, int lane) {
  BFU u;
  const unsigned int* p = fragLDS + f * 256 + lane * 8;
#pragma unroll
  for (int i = 0; i < 8; ++i) u.d[i] = p[i];
  return u.v;
}

// Gather one 16x32 bf16 A-fragment from a row-major f32 matrix.
// A layout: element e of lane l -> K = (e/8)*16 + (l/16)*8 + (e%8).
__device__ __forceinline__ v16bf gather_a(const float* row, int lhalf) {
  BFU u;
  const float* a0 = row + lhalf * 8;        // e = 0..7  -> K = lhalf*8 + e
  const float* b0 = row + 16 + lhalf * 8;   // e = 8..15 -> K = 16 + lhalf*8 + (e-8)
#pragma unroll
  for (int i = 0; i < 8; ++i) u.h[i] = (__bf16)a0[i];
#pragma unroll
  for (int i = 0; i < 8; ++i) u.h[8 + i] = (__bf16)b0[i];
  return u.v;
}

__global__ __launch_bounds__(128)
__attribute__((amdgpu_waves_per_eu(3)))   // floor; actual occupancy set by allocation
void tpa_kernel(
    const float* __restrict__ tptr, const float* __restrict__ zptr,
    const float* __restrict__ maskp,
    const float* __restrict__ Wq, const float* __restrict__ Wk,
    const float* __restrict__ Wv, const float* __restrict__ Wo,
    const float* __restrict__ bo, float* __restrict__ out) {
  __shared__ unsigned int fragLDS[NFRAGS * 256];   // 48 KB: bf16 B fragments
  __shared__ __bf16 oScr[4][16 * CT];              // 8 KB: per-wave O transpose

  const int lane  = threadIdx.x & 31;
  const int wv    = threadIdx.x >> 5;
  const int lhalf = lane >> 4;
  const int lcol  = lane & 15;

  // ---- Stage all weights into LDS as pre-swizzled bf16 B fragments (amortized
  // over 8 tiles per wave) ----
  for (int idx = threadIdx.x; idx < NFRAGS * 256; idx += blockDim.x) {
    int f    = idx >> 8;
    int rem  = idx & 255;
    int ln   = rem >> 3;
    int d    = rem & 7;
    int hh   = ln >> 4;
    int cc   = ln & 15;
    int e0 = 2 * d, e1 = 2 * d + 1;
    int k0 = ((e0 >> 3) << 4) + (hh << 3) + (e0 & 7);
    int k1 = ((e1 >> 3) << 4) + (hh << 3) + (e1 & 7);
    const float* W; int kbase, col, ncols;
    if (f < 16)      { int g = f;      W = Wq; kbase = (g >> 2) * 32; col = (g & 3) * 16 + cc; ncols = 64;  }
    else if (f < 24) { int g = f - 16; W = Wk; kbase = (g >> 2) * 32; col = (g & 3) * 16 + cc; ncols = 64;  }
    else if (f < 32) { int g = f - 24; W = Wv; kbase = (g >> 2) * 32; col = (g & 3) * 16 + cc; ncols = 64;  }
    else             { int g = f - 32; W = Wo; kbase = (g >> 3) * 32; col = (g & 7) * 16 + cc; ncols = 128; }
    float v0 = W[(size_t)(kbase + k0) * ncols + col];
    float v1 = W[(size_t)(kbase + k1) * ncols + col];
    fragLDS[idx] = pack2bf(v0, v1);
  }
  __syncthreads();

  const unsigned int gw = blockIdx.x * (blockDim.x >> 5) + wv;
  const unsigned int nw = gridDim.x * (blockDim.x >> 5);
  __bf16* scr = oScr[wv];

  for (unsigned int tile = gw; tile < NTILES; tile += nw) {
    const int pbase = tile * 16;
    const int prowA = pbase + lcol;   // A-fragment role: lane holds pixel row l%16

    // Prefetch next tile's z stream into L2.
    if (tile + nw < NTILES) {
      const int np = (tile + nw) * 16 + lcol;
      __builtin_prefetch(zptr + (size_t)np * CZ, 0, 1);
    }

    // ---- Q = Z * Wq, scaled into log2 domain (fully unrolled: static indices) ----
    const float* zrow = zptr + (size_t)prowA * CZ;
    v16bf aZ[4];
#pragma unroll
    for (int c = 0; c < 4; ++c) aZ[c] = gather_a(zrow + c * 32, lhalf);
    v8f q[4];
#pragma unroll
    for (int n = 0; n < 4; ++n) {
      v8f acc = {0.f, 0.f, 0.f, 0.f, 0.f, 0.f, 0.f, 0.f};
#pragma unroll
      for (int c = 0; c < 4; ++c)
        acc = wmma_bf16(aZ[c], load_bfrag(fragLDS, FRAG_WQ(c, n), lane), acc);
      q[n] = acc * (0.25f * LOG2E);  // 1/sqrt(C_HID) * log2(e)
    }
    __builtin_amdgcn_sched_barrier(0);   // don't interleave Q stage with attention

    // ---- Streaming attention over templates (T=4, exp2-domain softmax) ----
    float s_[NHEADS][8], o_[NHEADS][8];
#pragma unroll
    for (int h = 0; h < NHEADS; ++h)
#pragma unroll
      for (int r = 0; r < 8; ++r) { s_[h][r] = 0.f; o_[h][r] = 0.f; }

#pragma unroll 1   // rolled: keeps live set small
    for (int tt = 0; tt < NTEMPL; ++tt) {
      const float* trow = tptr + ((size_t)tt * PIXELS + prowA) * CT;
      // Prefetch the next template's rows while this one computes.
      if (tt + 1 < NTEMPL)
        __builtin_prefetch(trow + PIXELS * CT, 0, 1);

      const float bias = INF_F * (maskp[tt] - 1.0f) * LOG2E;  // uniform scalar
      v16bf aT[2];
#pragma unroll
      for (int c = 0; c < 2; ++c) aT[c] = gather_a(trow + c * 32, lhalf);

#pragma unroll     // full unroll (static h) + sched barriers to cap live window
      for (int h = 0; h < NHEADS; ++h) {
        v8f kd = {0.f, 0.f, 0.f, 0.f, 0.f, 0.f, 0.f, 0.f};
        v8f vd = {0.f, 0.f, 0.f, 0.f, 0.f, 0.f, 0.f, 0.f};
#pragma unroll
        for (int c = 0; c < 2; ++c) {
          kd = wmma_bf16(aT[c], load_bfrag(fragLDS, FRAG_WK(c, h), lane), kd);
          vd = wmma_bf16(aT[c], load_bfrag(fragLDS, FRAG_WV(c, h), lane), vd);
        }
#pragma unroll
        for (int r = 0; r < 8; ++r) {
          // logit (log2 domain): butterfly-reduce q*k over head h's 16 channels
          // (xor offsets 1,2,4,8 stay inside the 16-lane group).
          float pr = q[h][r] * kd[r];
          pr += __shfl_xor(pr, 1);
          pr += __shfl_xor(pr, 2);
          pr += __shfl_xor(pr, 4);
          pr += __shfl_xor(pr, 8);
          float w = exp2f(pr + bias);
          s_[h][r] += w;
          o_[h][r] += w * vd[r];
        }
        if (h == 1) __builtin_amdgcn_sched_barrier(0);  // <=2 head chains in flight
      }
    }

    // ---- Normalize and transpose O (C layout -> A layout) through LDS ----
#pragma unroll
    for (int h = 0; h < NHEADS; ++h)
#pragma unroll
      for (int r = 0; r < 8; ++r) {
        float val = o_[h][r] * __builtin_amdgcn_rcpf(s_[h][r]);
        int pix = r + 8 * lhalf;                 // C/D layout row
        scr[pix * CT + h * 16 + lcol] = (__bf16)val;
      }
    asm volatile("s_wait_dscnt 0x0" ::: "memory");  // same-wave LDS RAW fence

    v16bf aO[2];
#pragma unroll
    for (int c = 0; c < 2; ++c) {
      BFU u;
      const __bf16* base = scr + lcol * CT + c * 32 + lhalf * 8;
      const unsigned int* p0 = (const unsigned int*)(base);       // K = +0..7
      const unsigned int* p1 = (const unsigned int*)(base + 16);  // K = +16..23
#pragma unroll
      for (int i = 0; i < 4; ++i) u.d[i] = p0[i];
#pragma unroll
      for (int i = 0; i < 4; ++i) u.d[4 + i] = p1[i];
      aO[c] = u.v;
    }

    // ---- Out = O * Wo + bo (single base pointer -> immediate-offset stores) ----
    float* obase = out + (size_t)(pbase + 8 * lhalf) * CZ + lcol;
#pragma unroll
    for (int n = 0; n < 8; ++n) {
      v8f acc = {0.f, 0.f, 0.f, 0.f, 0.f, 0.f, 0.f, 0.f};
#pragma unroll
      for (int c = 0; c < 2; ++c)
        acc = wmma_bf16(aO[c], load_bfrag(fragLDS, FRAG_WO(c, n), lane), acc);
      const float bob = bo[n * 16 + lcol];   // L2/L0-resident, no persistent reg
#pragma unroll
      for (int r = 0; r < 8; ++r)
        obase[r * CZ + n * 16] = acc[r] + bob;
    }
  }
}

extern "C" void kernel_launch(void* const* d_in, const int* in_sizes, int n_in,
                              void* d_out, int out_size, void* d_ws, size_t ws_size,
                              hipStream_t stream) {
  (void)in_sizes; (void)n_in; (void)out_size; (void)d_ws; (void)ws_size;
  const float* t   = (const float*)d_in[0];
  const float* z   = (const float*)d_in[1];
  const float* msk = (const float*)d_in[2];
  const float* Wq  = (const float*)d_in[3];
  const float* Wk  = (const float*)d_in[4];
  const float* Wv  = (const float*)d_in[5];
  const float* Wo  = (const float*)d_in[6];
  const float* bo  = (const float*)d_in[7];
  float* out = (float*)d_out;

  // 16384 tiles of 16 pixels; 512 blocks * 4 waves = 2048 waves -> 8 tiles/wave
  // (amortizes the weight-staging phase to ~4%).
  dim3 grid(512), block(128);
  hipLaunchKernelGGL(tpa_kernel, grid, block, 0, stream,
                     t, z, msk, Wq, Wk, Wv, Wo, bo, out);
}
